// Block_39144331936432
// MI455X (gfx1250) — compile-verified, matrix operand order checked by reading, not run on hardware
//
#include <hip/hip_runtime.h>
#include <hip/hip_bf16.h>
#include <math.h>
#include <stdint.h>

// ---------------- types / helpers ----------------
typedef __attribute__((ext_vector_type(16))) __bf16 v16bf;
typedef __attribute__((ext_vector_type(8)))  float  v8f;

struct U128x2 { uint4 lo; uint4 hi; };

static __device__ __forceinline__ __bf16 f2bf(float f) {
  unsigned int u = __float_as_uint(f);
  unsigned int r = (u + 0x7FFFu + ((u >> 16) & 1u)) >> 16;
  unsigned short s = (unsigned short)r;
  return __builtin_bit_cast(__bf16, s);
}

// Load one 16x32-bf16 WMMA operand fragment for this lane.
// Per CDNA5 ISA layout, each lane needs two contiguous 8-element runs at
// p (e0..7) and p+16 (e8..15); caller folds the lane-half (+8) into p.
static __device__ __forceinline__ v16bf load_frag(const __bf16* p) {
  U128x2 u;
  u.lo = *(const uint4*)(p);
  u.hi = *(const uint4*)(p + 16);
  return __builtin_bit_cast(v16bf, u);
}

static __device__ __forceinline__ v8f wmma_bf(v16bf a, v16bf b, v8f c) {
  return __builtin_amdgcn_wmma_f32_16x16x32_bf16(false, a, false, b, (short)0, c,
                                                 false, false);
}

// CDNA5 async copy: 16B per lane straight from global memory into LDS,
// tracked with ASYNCcnt (no VGPR data staging).
static __device__ __forceinline__ void async_load_b128(unsigned lds_off,
                                                       const void* gptr) {
  asm volatile("global_load_async_to_lds_b128 %0, %1, off"
               :: "v"(lds_off), "v"((unsigned long long)(uintptr_t)gptr)
               : "memory");
}
static __device__ __forceinline__ void wait_async_le3() {
  asm volatile("s_wait_asynccnt 0x3" ::: "memory");
}
static __device__ __forceinline__ void wait_async_0() {
  asm volatile("s_wait_asynccnt 0x0" ::: "memory");
}
static __device__ __forceinline__ unsigned lds_offset(const void* p) {
  // generic pointer to LDS: addr[31:0] is the workgroup-relative LDS offset
  return (unsigned)(uintptr_t)p;
}

// ---------------- weight convert + transpose: out[n*K+k] = bf16(in[k*N+n]) ---
__global__ void wt_kernel(const float* __restrict__ in, __bf16* __restrict__ out,
                          int K, int N) {
  int id = blockIdx.x * 256 + threadIdx.x;
  if (id >= K * N) return;
  int n = id / K, k = id - n * K;
  out[id] = f2bf(in[(size_t)k * N + n]);
}

// ---------------- LayerNorm over rows of length 768 -> bf16 ----------------
__global__ __launch_bounds__(256)
void ln_kernel(const float* __restrict__ x, const float* __restrict__ g,
               const float* __restrict__ bb, __bf16* __restrict__ out) {
  __shared__ float sred[256];
  const int row = blockIdx.x;
  const int t = threadIdx.x;
  const float* xr = x + (size_t)row * 768;
  float vals[3];
  float s = 0.f;
  for (int i = 0; i < 3; ++i) { vals[i] = xr[t + i * 256]; s += vals[i]; }
  sred[t] = s; __syncthreads();
  for (int o = 128; o > 0; o >>= 1) { if (t < o) sred[t] += sred[t + o]; __syncthreads(); }
  float mean = sred[0] * (1.f / 768.f);
  __syncthreads();
  float vs = 0.f;
  for (int i = 0; i < 3; ++i) { float d = vals[i] - mean; vs += d * d; }
  sred[t] = vs; __syncthreads();
  for (int o = 128; o > 0; o >>= 1) { if (t < o) sred[t] += sred[t + o]; __syncthreads(); }
  float inv = rsqrtf(sred[0] * (1.f / 768.f) + 1e-5f);
  for (int i = 0; i < 3; ++i) {
    int c = t + i * 256;
    out[(size_t)row * 768 + c] = f2bf((vals[i] - mean) * inv * g[c] + bb[c]);
  }
}

// ---------------- tiled WMMA GEMM: out = A(MxK,bf16) @ Bt(NxK,bf16)^T -------
// Double-buffered LDS staging via CDNA5 async-to-LDS copies; last K-step
// peeled so the hot loop is branch-free.
// mode 0: QKV scatter (q,k row-major per head; v transposed (b,h,d,n))
// mode 1/3: outF = acc + bias + resid  (fp32)
// mode 2: outBF = bf16(gelu(acc + bias))
#define BM 128
#define BN 64
#define BK 32
#define LDS_STRIDE (BK + 8)  // 40 bf16 = 80 B, breaks bank aliasing
__global__ __launch_bounds__(256)
void gemm_kernel(const __bf16* __restrict__ A, const __bf16* __restrict__ Bt,
                 const float* __restrict__ bias, const float* __restrict__ resid,
                 float* __restrict__ outF, __bf16* __restrict__ outBF,
                 __bf16* __restrict__ qb, __bf16* __restrict__ kb,
                 __bf16* __restrict__ vb,
                 int M, int K, int Nn, int mode) {
  __shared__ __bf16 As[2][BM][LDS_STRIDE];
  __shared__ __bf16 Bs[2][BN][LDS_STRIDE];
  const int t = threadIdx.x;
  const int w = t >> 5, lane = t & 31;
  const int lrow = lane & 15;
  const int hs = (lane >= 16) ? 8 : 0;
  const int m0 = blockIdx.y * BM;
  const int n0 = blockIdx.x * BN;

  v8f acc[4];
  for (int j = 0; j < 4; ++j)
    for (int r = 0; r < 8; ++r) acc[j][r] = 0.f;

  // Per-thread staging coordinates (fixed across K-steps).
  // A tile 128x32: two 8-bf16 chunks/thread; B tile 64x32: one chunk/thread.
  const int ca = (t & 3) * 8;        // chunk column (elements)
  const int ra0 = t >> 2;            // A rows 0..63
  const int ra1 = 64 + (t >> 2);     // A rows 64..127
  // Out-of-range M rows: clamp (garbage only feeds epilogue-discarded rows)
  const int ma0 = (m0 + ra0 < M) ? (m0 + ra0) : (M - 1);
  const int ma1 = (m0 + ra1 < M) ? (m0 + ra1) : (M - 1);
  const unsigned aoff0 = lds_offset(&As[0][ra0][ca]);
  const unsigned aoff1 = lds_offset(&As[0][ra1][ca]);
  const unsigned boff  = lds_offset(&Bs[0][t >> 2][ca]);
  const unsigned abuf = BM * LDS_STRIDE * 2;  // bytes per A buffer
  const unsigned bbuf = BN * LDS_STRIDE * 2;  // bytes per B buffer
  const __bf16* ap0 = A + (size_t)ma0 * K + ca;
  const __bf16* ap1 = A + (size_t)ma1 * K + ca;
  const __bf16* bp  = Bt + (size_t)(n0 + (t >> 2)) * K + ca;

  auto issue_tile = [&](int k0, int buf) {
    async_load_b128(aoff0 + buf * abuf, ap0 + k0);
    async_load_b128(aoff1 + buf * abuf, ap1 + k0);
    async_load_b128(boff + buf * bbuf, bp + k0);
  };
  auto compute_tile = [&](int buf) {
    v16bf a = load_frag(&As[buf][w * 16 + lrow][hs]);
#pragma unroll
    for (int j = 0; j < 4; ++j) {
      v16bf b = load_frag(&Bs[buf][j * 16 + lrow][hs]);
      acc[j] = wmma_bf(a, b, acc[j]);
    }
  };

  const int nk = K / BK;  // always >= 2 here
  issue_tile(0, 0);
  for (int i = 0; i < nk - 1; ++i) {
    const int buf = i & 1;
    issue_tile((i + 1) * BK, buf ^ 1);
    wait_async_le3();  // 3 in-flight from next tile; current tile complete
    __syncthreads();
    compute_tile(buf);
    __syncthreads();   // all waves done reading before async overwrites buf
  }
  wait_async_0();
  __syncthreads();
  compute_tile((nk - 1) & 1);

  // epilogue: element r of lane -> row m0+w*16+hs+r, col n0+j*16+lrow
#pragma unroll
  for (int j = 0; j < 4; ++j) {
    int n = n0 + j * 16 + lrow;
#pragma unroll
    for (int r = 0; r < 8; ++r) {
      int m = m0 + w * 16 + hs + r;
      if (m >= M) continue;
      float v = acc[j][r];
      if (mode == 0) {
        int which = n / 768, c = n - which * 768;
        int hh = c >> 6, d = c & 63;
        int b = m >> 10, tk = m & 1023;
        __bf16 bv = f2bf(v);
        size_t bh = (size_t)(b * 12 + hh);
        if (which == 0)      qb[(bh * 1024 + tk) * 64 + d] = bv;
        else if (which == 1) kb[(bh * 1024 + tk) * 64 + d] = bv;
        else                 vb[(bh * 64 + d) * 1024 + tk] = bv;
      } else if (mode == 2) {
        float z = v + bias[n];
        float ge = 0.5f * z * (1.f + erff(z * 0.70710678f));
        outBF[(size_t)m * Nn + n] = f2bf(ge);
      } else {
        outF[(size_t)m * Nn + n] = v + bias[n] + resid[(size_t)m * Nn + n];
      }
    }
  }
}

// ---------------- attention: block = (b, 16-query tile), loops 12 heads -----
__global__ __launch_bounds__(256)
void attn_kernel(const __bf16* __restrict__ qb, const __bf16* __restrict__ kb,
                 const __bf16* __restrict__ vb, const float* __restrict__ amask,
                 __bf16* __restrict__ xatt, float* __restrict__ cls_out) {
  extern __shared__ char smem[];
  float*  S      = (float*)smem;               // 16*1024 fp32
  float*  Msk    = S + 16 * 1024;              // 16*1024 fp32
  __bf16* Abf    = (__bf16*)(Msk + 16 * 1024); // 16*1024 bf16
  float*  clsacc = (float*)(Abf + 16 * 1024);  // 1024
  float*  red    = clsacc + 1024;              // 256
  float*  rowv   = red + 256;                  // 16

  const int t = threadIdx.x, w = t >> 5, lane = t & 31;
  const int lrow = lane & 15;
  const int hs = (lane >= 16) ? 8 : 0;
  const int b = blockIdx.y;
  const int n0 = blockIdx.x * 16;

  // mask tile (head-invariant) streamed straight into LDS via async copies
  const float* mrow = amask + ((size_t)b * 1024 + n0) * 1024;
  const unsigned moff = lds_offset(Msk);
#pragma unroll
  for (int i = 0; i < 16; ++i) {
    int cc = t + i * 256;                 // float4 chunk index
    async_load_b128(moff + cc * 16, mrow + cc * 4);
  }
  for (int i = t; i < 1024; i += 256) clsacc[i] = 0.f;
  wait_async_0();
  __syncthreads();

  const float scale = 0.125f;  // 64^-0.5
  for (int h = 0; h < 12; ++h) {
    const __bf16* qbase = qb + (((size_t)(b * 12 + h)) * 1024 + n0) * 64;
    const __bf16* kbase = kb + (((size_t)(b * 12 + h)) * 1024) * 64;
    const __bf16* vbase = vb + (((size_t)(b * 12 + h)) * 64) * 1024;

    // Q fragments for both 32-wide K-steps of d=64
    v16bf qa0 = load_frag(qbase + lrow * 64 + hs);
    v16bf qa1 = load_frag(qbase + lrow * 64 + 32 + hs);

    // ---- S = scale * q @ k^T : each wave owns 8 column tiles of 16 ----
    for (int i = 0; i < 8; ++i) {
      int m0c = (w * 8 + i) * 16;
      v8f sacc; for (int r = 0; r < 8; ++r) sacc[r] = 0.f;
      const __bf16* kp = kbase + (size_t)(m0c + lrow) * 64;
      sacc = wmma_bf(qa0, load_frag(kp + hs), sacc);
      sacc = wmma_bf(qa1, load_frag(kp + 32 + hs), sacc);
      for (int r = 0; r < 8; ++r) S[(hs + r) * 1024 + m0c + lrow] = sacc[r] * scale;
    }
    __syncthreads();

    // ---- softmax (reference normalization) over each 1024-row ----
    const int sr = t >> 4, sc = t & 15;
    float lmax = -3.0e38f;
    for (int i = 0; i < 64; ++i) lmax = fmaxf(lmax, S[sr * 1024 + sc + 16 * i]);
    red[sr * 16 + sc] = lmax; __syncthreads();
    if (sc == 0) {
      float mx = red[sr * 16];
      for (int i = 1; i < 16; ++i) mx = fmaxf(mx, red[sr * 16 + i]);
      rowv[sr] = mx;
    }
    __syncthreads();
    float rmax = rowv[sr];
    float lsum = 0.f;
    for (int i = 0; i < 64; ++i) {
      int c = sc + 16 * i;
      float e = __expf(S[sr * 1024 + c] - rmax) * Msk[sr * 1024 + c];
      S[sr * 1024 + c] = e; lsum += e;
    }
    __syncthreads();
    red[sr * 16 + sc] = lsum; __syncthreads();
    if (sc == 0) {
      float s = 0.f;
      for (int i = 0; i < 16; ++i) s += red[sr * 16 + i];
      rowv[sr] = 1.f / (s + 1e-6f);
    }
    __syncthreads();
    float inv = rowv[sr];
    const float epsN = 1e-6f / 1024.f;
    for (int i = 0; i < 64; ++i) {
      int c = sc + 16 * i;
      float a = (S[sr * 1024 + c] + epsN) * inv;
      Abf[sr * 1024 + c] = f2bf(a);
      if (n0 == 0 && sr == 0 && c >= 1) clsacc[c] += a;   // cls_attn head-sum
    }
    __syncthreads();

    // ---- x_att tile = attn @ V : waves 0..3 own d-subtiles of 16 ----
    if (w < 4) {
      v8f a0, a1;
      for (int r = 0; r < 8; ++r) { a0[r] = 0.f; a1[r] = 0.f; }
      const __bf16* vrow = vbase + (size_t)(w * 16 + lrow) * 1024;
      for (int kk = 0; kk < 32; kk += 2) {
        a0 = wmma_bf(load_frag(&Abf[lrow * 1024 + kk * 32 + hs]),
                     load_frag(vrow + kk * 32 + hs), a0);
        a1 = wmma_bf(load_frag(&Abf[lrow * 1024 + (kk + 1) * 32 + hs]),
                     load_frag(vrow + (kk + 1) * 32 + hs), a1);
      }
      for (int r = 0; r < 8; ++r) {
        int n = n0 + hs + r;
        xatt[((size_t)(b * 1024) + n) * 768 + h * 64 + w * 16 + lrow] =
            f2bf(a0[r] + a1[r]);
      }
    }
    __syncthreads();
  }

  if (n0 == 0) {
    for (int i = t; i < 1024; i += 256)
      if (i >= 1) cls_out[(size_t)b * 1023 + (i - 1)] = clsacc[i] * (1.f / 12.f);
  }
}

// ---------------- top-k by iterative argmax (ties -> lower index) -----------
__global__ __launch_bounds__(256)
void topk_kernel(const float* __restrict__ cls, int off, int count, int k,
                 int* __restrict__ idx_i, float* __restrict__ idx_f) {
  __shared__ float vals[768];
  __shared__ float rv[256];
  __shared__ int   ri[256];
  const int b = blockIdx.x, t = threadIdx.x;
  const float* src = cls + (size_t)b * 1023 + off;
  for (int i = t; i < count; i += 256) vals[i] = src[i];
  __syncthreads();
  for (int it = 0; it < k; ++it) {
    float bv = -3.0e38f; int bi = 0x7FFFFFFF;
    for (int i = t; i < count; i += 256) {
      float v = vals[i];
      if (v > bv) { bv = v; bi = i; }
    }
    rv[t] = bv; ri[t] = bi; __syncthreads();
    for (int o = 128; o > 0; o >>= 1) {
      if (t < o) {
        if (rv[t + o] > rv[t] || (rv[t + o] == rv[t] && ri[t + o] < ri[t])) {
          rv[t] = rv[t + o]; ri[t] = ri[t + o];
        }
      }
      __syncthreads();
    }
    if (t == 0) {
      int win = ri[0];
      idx_i[b * k + it] = win;
      idx_f[b * k + it] = (float)win;
      vals[win] = -3.0e38f;
    }
    __syncthreads();
  }
}

// ---------------- gather pruned tokens: x_new = concat(cls, x[idx_a], x[idx_m])
__global__ void gather_kernel(const float* __restrict__ x1,
                              const int* __restrict__ ia,
                              const int* __restrict__ im,
                              float* __restrict__ xnew) {
  int id = blockIdx.x * 256 + threadIdx.x;
  if (id >= 8 * 513 * 768) return;
  int c = id % 768;
  int rr = id / 768;
  int r = rr % 513, b = rr / 513;
  int src;
  if (r == 0)       src = 0;
  else if (r <= 128) src = 1 + ia[b * 128 + (r - 1)];
  else               src = 256 + im[b * 384 + (r - 129)];
  xnew[id] = x1[((size_t)b * 1024 + src) * 768 + c];
}

__global__ void fill_kernel(float* __restrict__ p, int count, float v) {
  int id = blockIdx.x * 256 + threadIdx.x;
  if (id < count) p[id] = v;
}

// ---------------- host launcher ----------------
extern "C" void kernel_launch(void* const* d_in, const int* in_sizes, int n_in,
                              void* d_out, int out_size, void* d_ws, size_t ws_size,
                              hipStream_t stream) {
  (void)in_sizes; (void)n_in; (void)out_size; (void)ws_size;
  const float* x      = (const float*)d_in[0];
  const float* amask  = (const float*)d_in[1];
  const float* w_qkv  = (const float*)d_in[4];
  const float* w_proj = (const float*)d_in[5];
  const float* b_proj = (const float*)d_in[6];
  const float* g1     = (const float*)d_in[7];
  const float* b1     = (const float*)d_in[8];
  const float* g2     = (const float*)d_in[9];
  const float* b2     = (const float*)d_in[10];
  const float* w_fc1  = (const float*)d_in[11];
  const float* b_fc1  = (const float*)d_in[12];
  const float* w_fc2  = (const float*)d_in[13];
  const float* b_fc2  = (const float*)d_in[14];
  float* out = (float*)d_out;

  // output layout: x(8*513*768) | idx_a(1024) | idx_m(3072) | mask(8*513*513)
  const size_t OX  = 0;
  const size_t OIA = 8u * 513u * 768u;          // 3151872
  const size_t OIM = OIA + 8u * 128u;           // 3152896
  const size_t OMK = OIM + 8u * 384u;           // 3155968
  const int MASK_N = 8 * 513 * 513;             // 2105352

  size_t off = 0;
  char* base = (char*)d_ws;
  auto alloc = [&](size_t bytes) -> char* {
    char* p = base + off;
    off += (bytes + 255) & ~(size_t)255;
    return p;
  };
  __bf16* wqkvT = (__bf16*)alloc((size_t)2304 * 768 * 2);
  __bf16* wprojT = (__bf16*)alloc((size_t)768 * 768 * 2);
  __bf16* wfc1T = (__bf16*)alloc((size_t)3072 * 768 * 2);
  __bf16* wfc2T = (__bf16*)alloc((size_t)768 * 3072 * 2);
  __bf16* xn_bf = (__bf16*)alloc((size_t)8192 * 768 * 2);
  __bf16* qb    = (__bf16*)alloc((size_t)8 * 12 * 1024 * 64 * 2);
  __bf16* kb    = (__bf16*)alloc((size_t)8 * 12 * 1024 * 64 * 2);
  __bf16* vb    = (__bf16*)alloc((size_t)8 * 12 * 1024 * 64 * 2);
  __bf16* xatt  = (__bf16*)alloc((size_t)8192 * 768 * 2);
  float*  x1    = (float*)alloc((size_t)8192 * 768 * 4);
  float*  cls   = (float*)alloc((size_t)8 * 1023 * 4);
  int*    idxa  = (int*)alloc((size_t)8 * 128 * 4);
  int*    idxm  = (int*)alloc((size_t)8 * 384 * 4);
  float*  xnew  = (float*)alloc((size_t)8 * 513 * 768 * 4);
  __bf16* xn2   = (__bf16*)alloc((size_t)4104 * 768 * 2);
  __bf16* hbuf  = (__bf16*)alloc((size_t)4104 * 3072 * 2);

  // 1) weights -> bf16, transposed to (N,K)
  wt_kernel<<<(2304 * 768 + 255) / 256, 256, 0, stream>>>(w_qkv, wqkvT, 768, 2304);
  wt_kernel<<<(768 * 768 + 255) / 256, 256, 0, stream>>>(w_proj, wprojT, 768, 768);
  wt_kernel<<<(3072 * 768 + 255) / 256, 256, 0, stream>>>(w_fc1, wfc1T, 768, 3072);
  wt_kernel<<<(768 * 3072 + 255) / 256, 256, 0, stream>>>(w_fc2, wfc2T, 3072, 768);

  // 2) LN1
  ln_kernel<<<8192, 256, 0, stream>>>(x, g1, b1, xn_bf);

  // 3) QKV GEMM with head-scatter epilogue
  gemm_kernel<<<dim3(2304 / BN, 8192 / BM), 256, 0, stream>>>(
      xn_bf, wqkvT, nullptr, nullptr, nullptr, nullptr, qb, kb, vb,
      8192, 768, 2304, 0);

  // 4) attention (S, softmax, attn@V, cls_attn)
  size_t shmem = 16 * 1024 * 4      /* S */
               + 16 * 1024 * 4      /* Msk */
               + 16 * 1024 * 2      /* Abf */
               + 1024 * 4           /* clsacc */
               + 256 * 4 + 16 * 4;  /* red + rowv */
  attn_kernel<<<dim3(64, 8), 256, shmem, stream>>>(qb, kb, vb, amask, xatt, cls);

  // 5) proj GEMM + bias + residual -> x1
  gemm_kernel<<<dim3(768 / BN, 8192 / BM), 256, 0, stream>>>(
      xatt, wprojT, b_proj, x, x1, nullptr, nullptr, nullptr, nullptr,
      8192, 768, 768, 1);

  // 6) top-k (agent: 128 of 255, map: 384 of 768) + float idx outputs
  topk_kernel<<<8, 256, 0, stream>>>(cls, 0, 255, 128, idxa, out + OIA);
  topk_kernel<<<8, 256, 0, stream>>>(cls, 255, 768, 384, idxm, out + OIM);

  // 7) gather pruned tokens
  gather_kernel<<<(8 * 513 * 768 + 255) / 256, 256, 0, stream>>>(x1, idxa, idxm, xnew);

  // 8) LN2 + MLP
  ln_kernel<<<4104, 256, 0, stream>>>(xnew, g2, b2, xn2);
  gemm_kernel<<<dim3(3072 / BN, (4104 + BM - 1) / BM), 256, 0, stream>>>(
      xn2, wfc1T, b_fc1, nullptr, nullptr, hbuf, nullptr, nullptr, nullptr,
      4104, 768, 3072, 2);
  gemm_kernel<<<dim3(768 / BN, (4104 + BM - 1) / BM), 256, 0, stream>>>(
      hbuf, wfc2T, b_fc2, xnew, out + OX, nullptr, nullptr, nullptr, nullptr,
      4104, 3072, 768, 3);

  // 9) new_mask is provably all-ones (kAa==pa width, kAm==pm width)
  fill_kernel<<<(MASK_N + 255) / 256, 256, 0, stream>>>(out + OMK, MASK_N, 1.0f);
}
